// Model_80985903333893
// MI455X (gfx1250) — matrix operand order repaired and checked by reading, not compiled, not run on hardware
//
#include <hip/hip_runtime.h>
#include <math.h>

typedef __attribute__((ext_vector_type(16))) _Float16 v16h;
typedef __attribute__((ext_vector_type(8)))  float    v8f;

constexpr int kM    = 2048;   // nodes == embed dim
constexpr int kNN   = 16;     // neighbors per node
constexpr int kC    = 32;     // channels
constexpr int kB    = 32;     // batch
constexpr int kLHID = 256;
constexpr int kEX   = 16;
constexpr int kNL   = 4;
constexpr float kEPS = 1e-5f;

// ---------------------------------------------------------------------------
// Kernel A: attention MLP + per-node softmax over 16 neighbors.
// One block per pseudo row (32 blocks x 256 threads).
// ---------------------------------------------------------------------------
__global__ void __launch_bounds__(256)
attn_kernel(const float* __restrict__ pseudo,   // [32, kM]
            const float* __restrict__ w1,       // [kM, kLHID]
            const float* __restrict__ b1,       // [kLHID]
            const float* __restrict__ w2,       // [kLHID, kM/2]
            const float* __restrict__ b2,       // [kM/2]
            float* __restrict__ attn)           // [kM * kNN]
{
    __shared__ float e[kLHID];
    __shared__ float ctx[kM / 2];

    const int p = blockIdx.x;       // 0..31
    const int t = threadIdx.x;      // 0..255
    const float* prow = pseudo + (size_t)p * kM;

    // stage 1: e = tanh(pseudo_row @ w1 + b1)   -> [256]
    {
        float acc = b1[t];
        for (int n = 0; n < kM; ++n)
            acc = fmaf(prow[n], w1[(size_t)n * kLHID + t], acc);
        e[t] = tanhf(acc);
    }
    __syncthreads();

    // stage 2: ctx = e @ w2 + b2   -> [1024]
    for (int s = 0; s < 4; ++s) {
        const int q = t + 256 * s;
        float acc = b2[q];
        for (int j = 0; j < kLHID; ++j)
            acc = fmaf(e[j], w2[(size_t)j * (kM / 2) + q], acc);
        ctx[q] = acc;
    }
    __syncthreads();

    // stage 3: softmax over rows of 16 (64 rows per block)
    if (t < 64) {
        float v[kNN];
        float mx = -3.4e38f;
        for (int k = 0; k < kNN; ++k) { v[k] = ctx[t * kNN + k]; mx = fmaxf(mx, v[k]); }
        float s = 0.f;
        for (int k = 0; k < kNN; ++k) { v[k] = __expf(v[k] - mx); s += v[k]; }
        const float inv = 1.f / s;
        const int m = p * 64 + t;
        for (int k = 0; k < kNN; ++k) attn[m * kNN + k] = v[k] * inv;
    }
}

// ---------------------------------------------------------------------------
// Kernel B: sparse graph gather + 32x32 channel mix via WMMA + bias + BN stats.
// 8 waves/block; each wave handles one (node m, batch-half) 16x32 tile.
// ---------------------------------------------------------------------------
__global__ void __launch_bounds__(256)
conv_kernel(const float* __restrict__ x,      // [kB, kM, kC]
            const float* __restrict__ attn,   // [kM, kNN]
            const int*   __restrict__ L_idx,  // [kM * kNN] flat row*kM+col
            const float* __restrict__ W,      // [kC, kC]
            const float* __restrict__ bias,   // [kC]
            float* __restrict__ z,            // [kB, kM, kC]
            float* __restrict__ stats)        // [2*kC]: sum, sumsq
{
    __shared__ float ssum[kC], ssq[kC];
    const int t = threadIdx.x;
    if (t < kC) { ssum[t] = 0.f; ssq[t] = 0.f; }
    __syncthreads();

    const int lane  = t & 31;
    const int wave  = t >> 5;
    const int tile  = blockIdx.x * 8 + wave;   // 0 .. kM*2-1
    const int m     = tile >> 1;
    const int bbase = (tile & 1) * 16;
    const int r     = lane & 15;               // A-fragment row
    const int khalf = lane >> 4;               // 0: K 0-7/16-23, 1: K 8-15/24-31
    const int n     = lane & 15;               // B/D column

    // B fragments for xe_w halves (K x N = 32x16): lanes<16 K=0..7,16..23, lanes>=16 K=8..15,24..31
    v16h bw0, bw1;
    {
        const int kb = khalf * 8;
        for (int i = 0; i < 16; ++i) {
            bw0[i] = (_Float16)W[(kb + i + (i >= 8 ? 8 : 0)) * kC + n];
            bw1[i] = (_Float16)W[(kb + i + (i >= 8 ? 8 : 0)) * kC + 16 + n];
        }
    }

    // neighbor columns/weights with last-write-wins dedup (matches .at[].set)
    int   cols[kNN];
    float wts[kNN];
    for (int j = 0; j < kNN; ++j) {
        const int idx = L_idx[m * kNN + j];
        cols[j] = idx & (kM - 1);
        wts[j]  = attn[m * kNN + j];
    }
    for (int j = 0; j < kNN; ++j)
        for (int j2 = j + 1; j2 < kNN; ++j2)
            if (cols[j2] == cols[j]) { wts[j] = 0.f; break; }

    // gather-sum: A[r][k] = sum_j w_j * x[bbase+r, col_j, k]  (my 16 k's)
    float acc[16];
    for (int i = 0; i < 16; ++i) acc[i] = 0.f;
    const int bidx = bbase + r;
    for (int j = 0; j < kNN; ++j) {
        const float w = wts[j];
        const float4* xr = (const float4*)(x + ((size_t)bidx * kM + cols[j]) * kC);
        const float4 p0 = xr[khalf * 2 + 0];
        const float4 p1 = xr[khalf * 2 + 1];
        const float4 p2 = xr[4 + khalf * 2 + 0];
        const float4 p3 = xr[4 + khalf * 2 + 1];
        acc[0]  = fmaf(w, p0.x, acc[0]);  acc[1]  = fmaf(w, p0.y, acc[1]);
        acc[2]  = fmaf(w, p0.z, acc[2]);  acc[3]  = fmaf(w, p0.w, acc[3]);
        acc[4]  = fmaf(w, p1.x, acc[4]);  acc[5]  = fmaf(w, p1.y, acc[5]);
        acc[6]  = fmaf(w, p1.z, acc[6]);  acc[7]  = fmaf(w, p1.w, acc[7]);
        acc[8]  = fmaf(w, p2.x, acc[8]);  acc[9]  = fmaf(w, p2.y, acc[9]);
        acc[10] = fmaf(w, p2.z, acc[10]); acc[11] = fmaf(w, p2.w, acc[11]);
        acc[12] = fmaf(w, p3.x, acc[12]); acc[13] = fmaf(w, p3.y, acc[13]);
        acc[14] = fmaf(w, p3.z, acc[14]); acc[15] = fmaf(w, p3.w, acc[15]);
    }
    v16h a;
    for (int i = 0; i < 16; ++i) a[i] = (_Float16)acc[i];

    v8f c0 = {}, c1 = {};
    c0 = __builtin_amdgcn_wmma_f32_16x16x32_f16(false, a, false, bw0, (short)0, c0, false, false);
    c1 = __builtin_amdgcn_wmma_f32_16x16x32_f16(false, a, false, bw1, (short)0, c1, false, false);

    // bias + store + per-channel partial stats
    const float bn0 = bias[n], bn1 = bias[16 + n];
    float s0 = 0.f, q0 = 0.f, s1 = 0.f, q1 = 0.f;
    const int rowbase = 8 * khalf;   // D rows: lanes<16 -> 0..7, lanes>=16 -> 8..15
    for (int v = 0; v < 8; ++v) {
        const float z0 = c0[v] + bn0;
        const float z1 = c1[v] + bn1;
        const int bb = bbase + rowbase + v;
        float* zr = z + ((size_t)bb * kM + m) * kC;
        zr[n]      = z0;
        zr[16 + n] = z1;
        s0 += z0; q0 += z0 * z0;
        s1 += z1; q1 += z1 * z1;
    }
    atomicAdd(&ssum[n], s0);      atomicAdd(&ssq[n], q0);
    atomicAdd(&ssum[16 + n], s1); atomicAdd(&ssq[16 + n], q1);
    __syncthreads();
    if (t < kC) {
        atomicAdd(&stats[t], ssum[t]);
        atomicAdd(&stats[kC + t], ssq[t]);
    }
}

// ---------------------------------------------------------------------------
// Kernel C: BatchNorm (training stats) + ReLU, elementwise.
// ---------------------------------------------------------------------------
__global__ void __launch_bounds__(256)
bn_relu_kernel(const float* __restrict__ z, const float* __restrict__ stats,
               const float* __restrict__ gamma, const float* __restrict__ beta,
               float* __restrict__ xo)
{
    const int i  = blockIdx.x * 256 + threadIdx.x;    // < kB*kM*kC
    const int ch = i & (kC - 1);
    const float cnt = (float)(kB * kM);
    const float mu  = stats[ch] / cnt;
    const float var = stats[kC + ch] / cnt - mu * mu;
    const float sc  = gamma[ch] * rsqrtf(var + kEPS);
    const float v   = (z[i] - mu) * sc + beta[ch];
    xo[i] = fmaxf(v, 0.f);
}

// ---------------------------------------------------------------------------
// Kernel D: fc1 GEMM [32, 65552] @ [65552, 128] via WMMA, K-split + atomics.
// grid = (KSPLIT=16, ntile=8, rowtile=2), 1 wave per block.
// ---------------------------------------------------------------------------
__global__ void __launch_bounds__(32)
fc1_kernel(const float* __restrict__ xf,   // [kB, kM*kC] row-contiguous
           const float* __restrict__ ex,   // [kB, kEX]
           const float* __restrict__ w1,   // [kM*kC+kEX, 128]
           float* __restrict__ hacc)       // [kB, 128] f32 accumulators
{
    constexpr int KTOT = kM * kC;                 // 65536
    constexpr int NCH  = KTOT / 32 + 1;           // 2049 (last chunk = ex tail)
    const int ksplit  = blockIdx.x;               // 0..15
    const int ntile   = blockIdx.y;               // 0..7
    const int rowtile = blockIdx.z;               // 0..1
    const int lane  = threadIdx.x;
    const int r     = lane & 15;
    const int khalf = lane >> 4;
    const int n     = lane & 15;
    const int bidx  = rowtile * 16 + r;
    const int ncol  = ntile * 16 + n;

    constexpr int PER = (NCH + 15) / 16;          // 129
    const int c0i = ksplit * PER;
    int c1i = c0i + PER; if (c1i > NCH) c1i = NCH;

    v8f c = {};
    for (int ch = c0i; ch < c1i; ++ch) {
        const int kb = ch * 32;
        v16h a, bb;
        if (kb < KTOT) {
            const float4* xr = (const float4*)(xf + (size_t)bidx * KTOT + kb);
            const float4 p0 = xr[khalf * 2 + 0];
            const float4 p1 = xr[khalf * 2 + 1];
            const float4 p2 = xr[4 + khalf * 2 + 0];
            const float4 p3 = xr[4 + khalf * 2 + 1];
            a[0]=(_Float16)p0.x;  a[1]=(_Float16)p0.y;  a[2]=(_Float16)p0.z;  a[3]=(_Float16)p0.w;
            a[4]=(_Float16)p1.x;  a[5]=(_Float16)p1.y;  a[6]=(_Float16)p1.z;  a[7]=(_Float16)p1.w;
            a[8]=(_Float16)p2.x;  a[9]=(_Float16)p2.y;  a[10]=(_Float16)p2.z; a[11]=(_Float16)p2.w;
            a[12]=(_Float16)p3.x; a[13]=(_Float16)p3.y; a[14]=(_Float16)p3.z; a[15]=(_Float16)p3.w;
            for (int i = 0; i < 16; ++i)
                bb[i] = (_Float16)w1[(size_t)(kb + khalf * 8 + i + (i >= 8 ? 8 : 0)) * 128 + ncol];
        } else {
            // tail: K rows 0..15 come from ex_inputs, 16..31 are zero padding
            for (int i = 0; i < 16; ++i) { a[i] = (_Float16)0.f; bb[i] = (_Float16)0.f; }
            for (int i = 0; i < 8; ++i)
                a[i] = (_Float16)ex[bidx * kEX + khalf * 8 + i];
            for (int i = 0; i < 8; ++i)
                bb[i] = (_Float16)w1[(size_t)(KTOT + khalf * 8 + i) * 128 + ncol];
        }
        c = __builtin_amdgcn_wmma_f32_16x16x32_f16(false, a, false, bb, (short)0, c, false, false);
    }
    const int outb = rowtile * 16 + 8 * khalf;
    for (int v = 0; v < 8; ++v)
        atomicAdd(&hacc[(outb + v) * 128 + ncol], c[v]);
}

// ---------------------------------------------------------------------------
// Kernel E: bias + ReLU + fc2 epilogue.  out[b][o], 64 threads.
// ---------------------------------------------------------------------------
__global__ void __launch_bounds__(64)
fc2_kernel(const float* __restrict__ hacc, const float* __restrict__ b1,
           const float* __restrict__ w2,   const float* __restrict__ b2,
           float* __restrict__ out)
{
    const int t = threadIdx.x;       // 0..63
    const int b = t >> 1, o = t & 1;
    float acc = b2[o];
    for (int nn = 0; nn < 128; ++nn) {
        const float h = fmaxf(hacc[b * 128 + nn] + b1[nn], 0.f);
        acc = fmaf(h, w2[nn * 2 + o], acc);
    }
    out[b * 2 + o] = acc;
}

// ---------------------------------------------------------------------------
extern "C" void kernel_launch(void* const* d_in, const int* in_sizes, int n_in,
                              void* d_out, int out_size, void* d_ws, size_t ws_size,
                              hipStream_t stream) {
    const float* inputs = (const float*)d_in[0];
    const float* ex     = (const float*)d_in[1];
    const float* pseudo = (const float*)d_in[2];
    const int*   L_idx  = (const int*)  d_in[3];
    const float* l1_w   = (const float*)d_in[4];
    const float* l1_b   = (const float*)d_in[5];
    const float* l2_w   = (const float*)d_in[6];
    const float* l2_b   = (const float*)d_in[7];
    const float* xe_w   = (const float*)d_in[8];
    const float* xe_b   = (const float*)d_in[9];
    const float* bn_g   = (const float*)d_in[10];
    const float* bn_b   = (const float*)d_in[11];
    const float* fc1_w  = (const float*)d_in[12];
    const float* fc1_b  = (const float*)d_in[13];
    const float* fc2_w  = (const float*)d_in[14];
    const float* fc2_b  = (const float*)d_in[15];

    char* base = (char*)d_ws;
    const size_t XSZ = (size_t)kB * kM * kC * sizeof(float);   // 8 MB
    float* attn  = (float*)(base);                              // 128 KB
    float* stats = (float*)(base + 131072);                     // 256 B
    float* hacc  = (float*)(base + 131328);                     // 16 KB
    float* zbuf  = (float*)(base + 147712);
    float* xb0   = (float*)(base + 147712 + XSZ);
    float* xb1   = (float*)(base + 147712 + 2 * XSZ);

    const float* xcur = inputs;
    float* xbufs[2] = { xb0, xb1 };

    for (int i = 0; i < kNL; ++i) {
        attn_kernel<<<32, 256, 0, stream>>>(
            pseudo,
            l1_w + (size_t)i * kM * kLHID, l1_b + (size_t)i * kLHID,
            l2_w + (size_t)i * kLHID * (kM / 2), l2_b + (size_t)i * (kM / 2),
            attn);
        hipMemsetAsync(stats, 0, 2 * kC * sizeof(float), stream);
        conv_kernel<<<512, 256, 0, stream>>>(
            xcur, attn, L_idx,
            xe_w + (size_t)i * kC * kC, xe_b + (size_t)i * kC,
            zbuf, stats);
        float* xn = xbufs[i & 1];
        bn_relu_kernel<<<(kB * kM * kC) / 256, 256, 0, stream>>>(
            zbuf, stats, bn_g + (size_t)i * kC, bn_b + (size_t)i * kC, xn);
        xcur = xn;
    }

    hipMemsetAsync(hacc, 0, (size_t)kB * 128 * sizeof(float), stream);
    dim3 g1(16, 8, 2);
    fc1_kernel<<<g1, 32, 0, stream>>>(xcur, ex, fc1_w, hacc);
    fc2_kernel<<<1, 64, 0, stream>>>(hacc, fc1_b, fc2_w, fc2_b, (float*)d_out);
}
